// CrossGCNDense_44616120271595
// MI455X (gfx1250) — matrix operand order repaired and checked by reading, not compiled, not run on hardware
//
#include <hip/hip_runtime.h>
#include <stdint.h>

// ---------------- types for WMMA ----------------
typedef __bf16 bf16_t;
typedef bf16_t v16bf __attribute__((ext_vector_type(16)));
typedef float  v8f   __attribute__((ext_vector_type(8)));

union ABu { v16bf v; uint32_t d[8]; uint16_t h[16]; uint4 qd[2]; };
union Cu  { v8f v; float f[8]; };

__device__ __forceinline__ uint16_t f2bf(float x) {
  union { float f; uint32_t u; } a; a.f = x;
  uint32_t r = a.u + 0x7FFFu + ((a.u >> 16) & 1u);   // round-to-nearest-even
  return (uint16_t)(r >> 16);
}
__device__ __forceinline__ float bf2f(uint16_t h) {
  union { uint32_t u; float f; } a; a.u = ((uint32_t)h) << 16;
  return a.f;
}
// sum across the 16 lanes of this lane's half-wave (group-of-32 ds_swizzle, xor masks 1/2/4/8)
__device__ __forceinline__ float red16(float x) {
  x += __int_as_float(__builtin_amdgcn_ds_swizzle(__float_as_int(x), 0x041F));
  x += __int_as_float(__builtin_amdgcn_ds_swizzle(__float_as_int(x), 0x081F));
  x += __int_as_float(__builtin_amdgcn_ds_swizzle(__float_as_int(x), 0x101F));
  x += __int_as_float(__builtin_amdgcn_ds_swizzle(__float_as_int(x), 0x201F));
  return x;
}

// ---------------- problem constants ----------------
// B=2, N=512, G=4, P=32, F=64, L=256 ; NODES = B*N*G = 4096
// 16 nodes per workgroup -> 256 workgroups, 256 threads (8 wave32).

// LDS layout (bytes). Strides give 16B alignment for ds_load_b128 AND 16
// distinct banks across 16 consecutive rows (272B = 68 banks, 144B = 36 banks).
#define W1T_STR 136   // halves per row of W1^T [256][128]
#define SPB_STR 72    // halves per row of sp  [512][64]
#define OFF_W1T  0         // 69632 bytes (dead after phase 2)
#define OFF_SPB  69632     // 73728
#define OFF_QB   143360    // 2048
#define OFF_QF   145408    // 4096
#define OFF_B1   149504    // 1024
#define OFF_W2   150528    // 1024
#define OFF_WADJ 151552    // 2048
#define OFF_WF   153600    // 2048
#define OFF_INVQ 155648    // 64
#define SHMEM_BYTES 155712
// phases 3-6 alias the dead W1t region:
#define OFF_UF   0         // 4096 (written phase 4)
#define OFF_CRED 4096      // 4096 (zeroed phase 3)

__global__ __launch_bounds__(256) void crossgcn_kernel(
    const float* __restrict__ sp, const float* __restrict__ q,
    const float* __restrict__ W1, const float* __restrict__ b1,
    const float* __restrict__ W2, const float* __restrict__ b2,
    const float* __restrict__ Wg, const float* __restrict__ bg,
    float* __restrict__ out)
{
  extern __shared__ char smem[];
  uint16_t* W1t = (uint16_t*)(smem + OFF_W1T);  // [n(256)][k(128)] bf16, stride 136
  uint16_t* spb = (uint16_t*)(smem + OFF_SPB);  // [row(512)][f(64)] bf16, stride 72
  uint16_t* qb  = (uint16_t*)(smem + OFF_QB);   // [node(16)][k(64)] bf16
  float*    qf  = (float*)(smem + OFF_QF);      // [16][64] fp32
  float*    b1f = (float*)(smem + OFF_B1);      // [256]
  float*    W2f = (float*)(smem + OFF_W2);      // [256]
  float*    wadj= (float*)(smem + OFF_WADJ);    // [16][32] adjacency weights
  float*    wfv = (float*)(smem + OFF_WF);      // [16][32] w*inv_f
  float*    invq= (float*)(smem + OFF_INVQ);    // [16]
  float*    ufv = (float*)(smem + OFF_UF);      // [16][64] fp32 (aliases W1t)
  float*    Cred= (float*)(smem + OFF_CRED);    // [16][64] fp32 (aliases W1t)

  const int tid   = threadIdx.x;
  const int nb    = blockIdx.x * 16;       // first node of this tile
  const int wv    = tid >> 5;              // wave id 0..7
  const int lane  = tid & 31;
  const int lhalf = lane & 15;
  const int hi    = lane >> 4;             // 0 / 1

  // ---------------- Phase 1: stage operands into LDS (bf16, vectorized) ----------
  for (int idx4 = tid * 4; idx4 < 128 * 256; idx4 += 1024) {  // W1 [128][256] -> W1^T
    float4 w4 = *(const float4*)(W1 + idx4);
    int k = idx4 >> 8, n0 = idx4 & 255;                       // 4 cols, same k
    W1t[(n0 + 0) * W1T_STR + k] = f2bf(w4.x);
    W1t[(n0 + 1) * W1T_STR + k] = f2bf(w4.y);
    W1t[(n0 + 2) * W1T_STR + k] = f2bf(w4.z);
    W1t[(n0 + 3) * W1T_STR + k] = f2bf(w4.w);
  }
  const float* spg = sp + (size_t)nb * 32 * 64;
  for (int idx4 = tid * 4; idx4 < 512 * 64; idx4 += 1024) {   // sample points
    float4 s4 = *(const float4*)(spg + idx4);
    int row = idx4 >> 6, f = idx4 & 63;
    uint2 pk;
    pk.x = (uint32_t)f2bf(s4.x) | ((uint32_t)f2bf(s4.y) << 16);
    pk.y = (uint32_t)f2bf(s4.z) | ((uint32_t)f2bf(s4.w) << 16);
    *(uint2*)(spb + row * SPB_STR + f) = pk;                  // ds_store_b64
  }
  const float* qg = q + (size_t)nb * 64;
  {
    int idx4 = tid * 4;                                       // 1024 = 256*4 exactly
    float4 q4 = *(const float4*)(qg + idx4);
    *(float4*)(qf + idx4) = q4;
    uint2 pk;
    pk.x = (uint32_t)f2bf(q4.x) | ((uint32_t)f2bf(q4.y) << 16);
    pk.y = (uint32_t)f2bf(q4.z) | ((uint32_t)f2bf(q4.w) << 16);
    *(uint2*)(qb + idx4) = pk;
  }
  b1f[tid] = b1[tid];                                         // tid < 256 always
  W2f[tid] = W2[tid];
  __syncthreads();

  const float b2v = b2[0];

  // ---------------- Phase 2: edge MLP  h = cat@W1 ; LN ; leaky ; sigmoid(h@W2) ----
  // GEMM M=512 (16 nodes * 32 pts), N=256, K=128 (cols 0..63 = q, 64..127 = sp)
  // Flattened kt*nt loop with 1-deep B double-buffer: loads for tile i+1 issue
  // before the WMMA of tile i (s_wait_dscnt<=4, no WAR hazard NOPs), and
  // sched_barrier(0) pins the pipeline so live registers stay bounded.
  for (int mt = wv; mt < 32; mt += 8) {
    const int node = mt >> 1;
    const int kb   = hi ? 8 : 0;     // table-order K base (halves); unused
    const int kb2  = hi ? 16 : 0;    // K base (halves) of A and B: k = 16*hi + s
    const uint16_t* qrow = qb  + node * 64;
    const uint16_t* srow = spb + (mt * 16 + lhalf) * SPB_STR;

    Cu acc[16];
#pragma unroll
    for (int nt = 0; nt < 16; ++nt)
#pragma unroll
      for (int i = 0; i < 8; ++i) acc[nt].f[i] = 0.0f;

    ABu A[4];                        // K-tiles 0,1 from q; 2,3 from sp (2x b128 each)
#pragma unroll
    for (int kt = 0; kt < 4; ++kt) {
      const uint16_t* arow = ((kt < 2) ? (qrow + kt * 32) : (srow + (kt - 2) * 32)) + kb2;
      A[kt].qd[0] = *(const uint4*)(arow);
      A[kt].qd[1] = *(const uint4*)(arow + 8);
    }

    const uint16_t* wbase = W1t + lhalf * W1T_STR + kb2;
    ABu Bcur;
    Bcur.qd[0] = *(const uint4*)(wbase);
    Bcur.qd[1] = *(const uint4*)(wbase + 8);
#pragma unroll
    for (int i = 0; i < 64; ++i) {
      const int kt = i >> 4, nt = i & 15;
      ABu Bnext = Bcur;
      if (i < 63) {
        const int kt2 = (i + 1) >> 4, nt2 = (i + 1) & 15;
        const uint16_t* wrow = wbase + nt2 * 16 * W1T_STR + kt2 * 32;
        Bnext.qd[0] = *(const uint4*)(wrow);
        Bnext.qd[1] = *(const uint4*)(wrow + 8);
      }
      acc[nt].v = __builtin_amdgcn_wmma_f32_16x16x32_bf16(
          false, A[kt].v, false, Bcur.v, (short)0, acc[nt].v, false, false);
      __builtin_amdgcn_sched_barrier(0);      // keep pipeline shape / live set
      Bcur = Bnext;
    }

#pragma unroll
    for (int nt = 0; nt < 16; ++nt) {          // bias before LN stats
      const float bv = b1f[nt * 16 + lhalf];
#pragma unroll
      for (int v = 0; v < 8; ++v) acc[nt].f[v] += bv;
    }

    // per-row (row = v + 8*hi within tile): LayerNorm over 256 cols, leaky, dot W2
#pragma unroll
    for (int v = 0; v < 8; ++v) {
      float s = 0.0f, sq = 0.0f;
#pragma unroll
      for (int nt = 0; nt < 16; ++nt) { float x = acc[nt].f[v]; s += x; sq += x * x; }
      s  = red16(s);
      sq = red16(sq);
      const float mu  = s * (1.0f / 256.0f);
      const float var = sq * (1.0f / 256.0f) - mu * mu;
      const float rs  = rsqrtf(var + 1e-5f);
      float dot = 0.0f;
#pragma unroll
      for (int nt = 0; nt < 16; ++nt) {
        float x = (acc[nt].f[v] - mu) * rs;
        x = (x >= 0.0f) ? x : 0.01f * x;                     // LeakyReLU
        dot += x * W2f[nt * 16 + lhalf];
      }
      dot = red16(dot);
      const float wr = 1.0f / (1.0f + __expf(-(dot + b2v))); // sigmoid
      if (lhalf == v) {                                      // one writer per (v,hi)
        int p = (mt & 1) * 16 + v + 8 * hi;
        wadj[node * 32 + p] = wr;
      }
    }
  }
  __syncthreads();

  // ---------------- Phase 3: degree norms (+ zero Cred, aliases dead W1t) --------
  if (tid < 16) {
    float s = 0.0f;
    for (int p = 0; p < 32; ++p) s += wadj[tid * 32 + p];
    invq[tid] = rsqrtf(1.0f + s);
  }
  for (int i = tid; i < 512; i += 256)
    wfv[i] = wadj[i] * rsqrtf(1.0f + wadj[i]);               // w * inv_f
  for (int idx = tid; idx < 16 * 64; idx += 256) Cred[idx] = 0.0f;
  __syncthreads();

  // ---------------- Phase 4: u = sum_p (w*inv_f)_p * sp_p + inv_q * q ------------
  for (int idx = tid; idx < 16 * 64; idx += 256) {
    int n = idx >> 6, f = idx & 63;
    float s = 0.0f;
    for (int p = 0; p < 32; ++p)
      s += wfv[n * 32 + p] * bf2f(spb[(n * 32 + p) * SPB_STR + f]);
    ufv[idx] = s + invq[n] * qf[idx];
  }
  __syncthreads();

  // ---------------- Phase 5: T[16,64] = M[16,4160] @ [Wg ; bg]  (bf16 WMMA) ------
  // M[n, k*64+f] = q[n][k]*u[n][f]  (k=64 -> ones row picks up bg). K split across waves.
  {
    Cu acc2[4];
#pragma unroll
    for (int nt = 0; nt < 4; ++nt)
#pragma unroll
      for (int i = 0; i < 8; ++i) acc2[nt].f[i] = 0.0f;

    const int n   = lhalf;            // A-matrix row = node
    const int kb  = hi ? 8 : 0;
    const int kb2 = hi ? 16 : 0;
    for (int kt = wv; kt < 130; kt += 8) {
      ABu Av;                                        // build A on the fly (fp32 mul)
#pragma unroll
      for (int v = 0; v < 8; ++v) {
        int Koff = kb2 + 2 * v;
#pragma unroll
        for (int e2 = 0; e2 < 2; ++e2) {
          int Kp = kt * 32 + Koff + e2;
          int k = Kp >> 6, f = Kp & 63;
          float qv = (k < 64) ? qf[n * 64 + k] : 1.0f;
          Av.h[2 * v + e2] = f2bf(qv * ufv[n * 64 + f]);
        }
      }
#pragma unroll
      for (int nt = 0; nt < 4; ++nt) {
        ABu Bv;
        const int col  = nt * 16 + lhalf;
        const int Kpb  = kt * 32 + kb2;              // tile never straddles 4096
        const float* srcbase = (Kpb < 4096)
            ? (Wg + (size_t)Kpb * 64 + col)
            : (bg + (size_t)(Kpb - 4096) * 64 + col);
#pragma unroll
        for (int j = 0; j < 16; ++j)
          Bv.h[j] = f2bf(srcbase[(size_t)j * 64]);
        acc2[nt].v = __builtin_amdgcn_wmma_f32_16x16x32_bf16(
            false, Av.v, false, Bv.v, (short)0, acc2[nt].v, false, false);
      }
    }
    // cross-wave reduction of K-partials into LDS
#pragma unroll
    for (int nt = 0; nt < 4; ++nt)
#pragma unroll
      for (int v = 0; v < 8; ++v) {
        int node = v + 8 * hi;
        int e    = nt * 16 + lhalf;
        atomicAdd(&Cred[node * 64 + e], acc2[nt].f[v]);
      }
  }
  __syncthreads();

  // ---------------- Phase 6: out = q + inv_q * T ----------------
  float* outp = out + (size_t)nb * 64;
  {
    int idx4 = tid * 4;                                       // 1024 = 256*4 exactly
    int n = idx4 >> 6;
    float4 qv = *(const float4*)(qf + idx4);
    float4 cv = *(const float4*)(Cred + idx4);
    float iq = invq[n];
    float4 o;
    o.x = qv.x + iq * cv.x;  o.y = qv.y + iq * cv.y;
    o.z = qv.z + iq * cv.z;  o.w = qv.w + iq * cv.w;
    *(float4*)(outp + idx4) = o;
  }
}

extern "C" void kernel_launch(void* const* d_in, const int* in_sizes, int n_in,
                              void* d_out, int out_size, void* d_ws, size_t ws_size,
                              hipStream_t stream) {
  (void)in_sizes; (void)n_in; (void)d_ws; (void)ws_size; (void)out_size;
  const float* sp  = (const float*)d_in[0];
  const float* q   = (const float*)d_in[1];
  const float* W1  = (const float*)d_in[2];
  const float* b1  = (const float*)d_in[3];
  const float* W2  = (const float*)d_in[4];
  const float* b2  = (const float*)d_in[5];
  const float* Wg  = (const float*)d_in[6];
  const float* bg  = (const float*)d_in[7];
  float* out = (float*)d_out;

  hipFuncSetAttribute((const void*)crossgcn_kernel,
                      hipFuncAttributeMaxDynamicSharedMemorySize, SHMEM_BYTES);
  // 4096 nodes / 16 per workgroup = 256 workgroups, 8 wave32 each
  crossgcn_kernel<<<256, 256, SHMEM_BYTES, stream>>>(sp, q, W1, b1, W2, b2, Wg, bg, out);
}
